// SphericalCNN_69458211111145
// MI455X (gfx1250) — matrix operand order repaired and checked by reading, not compiled, not run on hardware
//
#include <hip/hip_runtime.h>
#include <hip/hip_bf16.h>
#include <math.h>

// ---------------------------------------------------------------------------
// MI455X (gfx1250) spherical CNN pipeline, bf16 WMMA everywhere it matters.
//   - dominant cost: 2 fused SO(3)-grid activation transforms (33 GFLOP)
//     -> fused GEMM1/relu/GEMM2 kernel, grid intermediate never touches HBM
//   - act kernel double-buffers B fragments through LDS (8x less L2 traffic)
//   - so3_linear layers: per-l WMMA GEMMs (dims are multiples of 16/32)
//   - Wigner constants generated on device (closed-form small-d)
//   - all GEMM tile loops have compile-time trip counts (no VGPR movrel)
// ---------------------------------------------------------------------------

typedef __bf16 bf16_t;
typedef __attribute__((ext_vector_type(16))) __bf16 v16bf;
typedef __attribute__((ext_vector_type(8)))  float  v8f;
typedef __attribute__((ext_vector_type(4)))  unsigned int v4u;

union FragU { v16bf v; v4u u[2]; };

__device__ inline v8f vzero8() {
  v8f z;
#pragma unroll
  for (int i = 0; i < 8; ++i) z[i] = 0.f;
  return z;
}

__device__ inline v8f wmma_bf16(v16bf a, v16bf b, v8f c) {
  return __builtin_amdgcn_wmma_f32_16x16x32_bf16(false, a, false, b, (short)0, c,
                                                 false, false);
}

// A fragment (16x32 bf16) from row-major [.., ld] at (row0, k0).
// CDNA5 A layout: lane<16 -> K[0..7]+K[16..23], lane>=16 -> K[8..15]+K[24..31],
// each run of 8 halves contiguous -> two b128 loads.
__device__ inline v16bf load_afrag(const bf16_t* base, int ld, int row0, int k0,
                                   int lane) {
  int r = row0 + (lane & 15);
  int hs = (lane >> 4) * 8;
  const bf16_t* p = base + (long)r * ld + k0;
  FragU f;
  f.u[0] = *(const v4u*)(p + hs);
  f.u[1] = *(const v4u*)(p + 16 + hs);
  return f.v;
}

// B fragment from pre-swizzled global buffer: 32 lanes x 16 halves contiguous.
__device__ inline v16bf load_bfrag(const bf16_t* buf, long fragIdx, int lane) {
  const bf16_t* p = buf + fragIdx * 512 + lane * 16;
  FragU f;
  f.u[0] = *(const v4u*)(p);
  f.u[1] = *(const v4u*)(p + 8);
  return f.v;
}

// B fragment from LDS staging (fragHalfOff = fragment offset in halves)
__device__ inline v16bf lds_bfrag(const bf16_t* buf, int fragHalfOff, int lane) {
  const bf16_t* p = buf + fragHalfOff + lane * 16;
  FragU f;
  f.u[0] = *(const v4u*)(p);
  f.u[1] = *(const v4u*)(p + 8);
  return f.v;
}

// ------------------------- Wigner-D constant generation ---------------------

__device__ inline float factf(int n) {
  float r = 1.f;
  for (int i = 2; i <= n; ++i) r *= (float)i;
  return r;
}

__device__ inline int imin_(int a, int b) { return a < b ? a : b; }
__device__ inline int imax_(int a, int b) { return a > b ? a : b; }

// standard Wigner small-d, d^l_{mp,m}(beta), cb2=cos(b/2) sb2=sin(b/2)
__device__ float small_d(int l, int mp, int m, float cb2, float sb2) {
  float pref = sqrtf(factf(l + mp) * factf(l - mp) * factf(l + m) * factf(l - m));
  int kmin = imax_(0, m - mp), kmax = imin_(l + m, l - mp);
  float sum = 0.f;
  for (int k = kmin; k <= kmax; ++k) {
    float denom = factf(l + m - k) * factf(k) * factf(mp - m + k) * factf(l - mp - k);
    float sgn = ((mp - m + k) & 1) ? -1.f : 1.f;
    float c = powf(cb2, (float)(2 * l + m - mp - 2 * k));
    float s = powf(sb2, (float)(mp - m + 2 * k));
    sum += sgn * c * s / denom;
  }
  return pref * sum;
}

// q(m, r): complex change-of-basis element (global (-i)^l phase cancels)
__device__ inline void qval(int m, int r, float* re, float* im) {
  const float is2 = 0.70710678118654752f;
  *re = 0.f; *im = 0.f;
  if (m == 0) { if (r == 0) *re = 1.f; return; }
  if (m < 0) {
    if (r == -m) *re = is2;
    else if (r == m) *im = -is2;
  } else {
    float s = (m & 1) ? -is2 : is2;
    if (r == m) *re = s;
    else if (r == -m) *im = s;
  }
}

// real-basis Wigner D entry; r1,r2 in [-l,l]; phases e^{+i m a}, e^{+i m g}
__device__ float wigner_real(int l, int r1, int r2, float al, float be, float ga) {
  float sb2, cb2;
  __sincosf(be * 0.5f, &sb2, &cb2);
  float res = 0.f;
  for (int t1 = 0; t1 < 2; ++t1) {
    if (t1 && r1 == 0) break;
    int m1 = t1 ? -r1 : r1;
    float q1r, q1i;
    qval(m1, r1, &q1r, &q1i);
    q1i = -q1i;  // conj
    for (int t2 = 0; t2 < 2; ++t2) {
      if (t2 && r2 == 0) break;
      int m2 = t2 ? -r2 : r2;
      float q2r, q2i;
      qval(m2, r2, &q2r, &q2i);
      float d = small_d(l, m1, m2, cb2, sb2);
      float sp, cp;
      __sincosf((float)m1 * al + (float)m2 * ga, &sp, &cp);
      float ar = q1r * q2r - q1i * q2i;
      float ai = q1r * q2i + q1i * q2r;
      res += d * (ar * cp - ai * sp);
    }
  }
  return res;
}

__constant__ int c_OX[5] = {0, 1, 10, 35, 84};  // cumulative d^2
#define PI_F 3.14159265358979323846f

// near-identity SO3 grid -> D_K bf16 [192, 176] (pads already zero)
__global__ void gen_DK(bf16_t* dk) {
  int p = threadIdx.x;
  if (p >= 192) return;
  int ia = p / 24, ib = (p / 8) % 3, ic = p % 8;
  float a = 2.f * PI_F * (float)ia / 8.f;
  float b = (float)(ib + 1) * (PI_F / 24.f);
  float g = (-2.f * PI_F + (float)ic * (4.f * PI_F / 7.f)) - a;
  for (int l = 0; l <= 4; ++l) {
    float sc = sqrtf((float)(2 * l + 1));
    int off = c_OX[l], d = 2 * l + 1;
    for (int r1 = -l; r1 <= l; ++r1)
      for (int r2 = -l; r2 <= l; ++r2)
        dk[(long)p * 176 + off + (r1 + l) * d + (r2 + l)] =
            (bf16_t)(sc * wigner_real(l, r1, r2, a, b, g));
  }
}

// full SO3 grid: DACT bf16 [4096,192], D2 = sqrt(2)*QW*DACT bf16 [4096,176]
__global__ void gen_DACT(bf16_t* dact, bf16_t* d2) {
  int t = blockIdx.x * blockDim.x + threadIdx.x;
  if (t >= 4096 * 5) return;
  int p = t / 5, l = t % 5;
  int ia = p >> 8, ib = (p >> 4) & 15, ic = p & 15;
  float a = 2.f * PI_F * (float)ia / 16.f;
  float b = ((float)ib + 0.5f) * (PI_F / 16.f);
  float g = 2.f * PI_F * (float)ic / 16.f;
  float S = 0.f;
  for (int j = 0; j < 16; ++j) S += __sinf(((float)j + 0.5f) * (PI_F / 16.f));
  float qw = __sinf(b) / (256.f * S);
  float sc = sqrtf((float)(2 * l + 1));
  float w2 = 1.41421356237f * qw;
  int off = c_OX[l], d = 2 * l + 1;
  for (int r1 = -l; r1 <= l; ++r1)
    for (int r2 = -l; r2 <= l; ++r2) {
      float v = sc * wigner_real(l, r1, r2, a, b, g);
      long i = off + (r1 + l) * d + (r2 + l);
      dact[(long)p * 192 + i] = (bf16_t)v;
      d2[(long)p * 176 + i] = (bf16_t)(w2 * v);
    }
}

// S2 grid SH: Y f32 [24, 9]
__global__ void gen_Y(float* y) {
  int p = threadIdx.x;
  if (p >= 24) return;
  int ia = p / 3, ib = p % 3;
  float a = 2.f * PI_F * (float)ia / 8.f;
  float b = (float)(ib + 1) * (PI_F / 24.f);
  const int soff[3] = {0, 1, 4};
  for (int l = 0; l <= 2; ++l) {
    float sc = sqrtf((float)(2 * l + 1));
    for (int r1 = -l; r1 <= l; ++r1)
      y[(long)p * 9 + soff[l] + (r1 + l)] = sc * wigner_real(l, r1, 0, a, b, 0.f);
  }
}

// -------------------------------- repacks -----------------------------------

// B-operand swizzle from row-major bf16 src (optionally transposed access).
// actOrder=1 reorders fragments as [nc=nt/2][kt][nt&1] (contiguous 12KB/chunk).
__global__ void repackB(const bf16_t* src, bf16_t* dst, int K, int Nvalid, int ld,
                        int KT, int NT, int transposed, int actOrder) {
  long t = (long)blockIdx.x * blockDim.x + threadIdx.x;
  long nfr = (long)KT * NT;
  if (t >= nfr * 32) return;
  int lane = (int)(t & 31);
  long frag = t >> 5;
  int nt = (int)(frag % NT), kt = (int)(frag / NT);
  long dstFrag = actOrder ? (((long)(nt >> 1) * KT + kt) * 2 + (nt & 1)) : frag;
  int hiK = (lane >= 16) ? 8 : 0;
  int n = nt * 16 + (lane & 15);
  bf16_t* out = dst + dstFrag * 512 + lane * 16;
  for (int h = 0; h < 16; ++h) {
    int k = kt * 32 + hiK + h + (h & 8);
    bf16_t v = (bf16_t)0.f;
    if (k < K && n < Nvalid)
      v = transposed ? src[(long)n * ld + k] : src[(long)k * ld + n];
    out[h] = v;
  }
}

// per-l psi -> B fragments: B[k=(x,u), n=(y,v)] = psi[(x*f_out+y), op + u*d + v]
__global__ void repackPsiB(const float* psi, int ldpsi, int opoff, int d, int f_out,
                           int KT, int NT, bf16_t* dst) {
  long t = (long)blockIdx.x * blockDim.x + threadIdx.x;
  long nfr = (long)KT * NT;
  if (t >= nfr * 32) return;
  int lane = (int)(t & 31);
  long frag = t >> 5;
  int nt = (int)(frag % NT), kt = (int)(frag / NT);
  int hiK = (lane >= 16) ? 8 : 0;
  int n = nt * 16 + (lane & 15);
  int y = n / d, v = n % d;
  bf16_t* out = dst + frag * 512 + lane * 16;
  for (int h = 0; h < 16; ++h) {
    int k = kt * 32 + hiK + h + (h & 8);
    int x = k / d, u = k % d;
    out[h] = (bf16_t)psi[(long)(x * f_out + y) * ldpsi + opoff + u * d + v];
  }
}

// per-l A gather: A[(b,m),(x,u)] = src[(b*f_in+x), ox + u*d + m]  (f32 or bf16)
__global__ void repackA_lin(const void* src, int srcF32, int ldx, int oxoff, int d,
                            int f_in, bf16_t* dst) {
  int M = 64 * d, K = f_in * d;
  long tot = (long)M * K;
  long t = (long)blockIdx.x * blockDim.x + threadIdx.x;
  if (t >= tot) return;
  int row = (int)(t / K), k = (int)(t % K);
  int b = row / d, m = row % d;
  int x = k / d, u = k % d;
  long si = (long)(b * f_in + x) * ldx + oxoff + u * d + m;
  float v = srcF32 ? ((const float*)src)[si] : (float)((const bf16_t*)src)[si];
  dst[t] = (bf16_t)v;
}

__global__ void cvt_f32_bf16(const float* s, bf16_t* d, long n) {
  long t = (long)blockIdx.x * blockDim.x + threadIdx.x;
  if (t < n) d[t] = (bf16_t)s[t];
}

// ------------------------------ generic GEMM --------------------------------
// C[M,N] = A[M,K] * B(frag), bf16 x bf16 -> f32.  8 waves, 128 rows/wg,
// exactly 4 n-tiles per wave (NT must be a multiple of 4; pad tiles are zero).
// mode 0: f32 row-major out.  mode 1: scatter bf16 into Xact.
__global__ __launch_bounds__(256) void wmma_gemm(
    const bf16_t* A, int lda, const bf16_t* Bfrag, int KT, int NT, float* Cout,
    int ldc, float scale, int M, int Nvalid, int mode, bf16_t* Xact, int ldxa,
    int oxoff, int d, int f_out) {
  int lane = threadIdx.x & 31, wave = threadIdx.x >> 5;
  int row0 = blockIdx.x * 128 + wave * 16;
  if (row0 >= M) return;
  int nt0 = blockIdx.y * 4;
  v8f acc[4];
#pragma unroll
  for (int t = 0; t < 4; ++t) acc[t] = vzero8();
  for (int kt = 0; kt < KT; ++kt) {
    v16bf a = load_afrag(A, lda, row0, kt * 32, lane);
    const bf16_t* bp = Bfrag + ((long)kt * NT + nt0) * 512;
    v16bf b[4];
#pragma unroll
    for (int t = 0; t < 4; ++t) b[t] = load_bfrag(bp, t, lane);
#pragma unroll
    for (int t = 0; t < 4; ++t) acc[t] = wmma_bf16(a, b[t], acc[t]);
  }
  int rbase = row0 + ((lane >> 4) << 3);
  int c0 = lane & 15;
#pragma unroll
  for (int t = 0; t < 4; ++t) {
    int n = (nt0 + t) * 16 + c0;
    if (n >= Nvalid) continue;
#pragma unroll
    for (int v = 0; v < 8; ++v) {
      float val = acc[t][v] * scale;
      int r = rbase + v;
      if (mode == 0) {
        Cout[(long)r * ldc + n] = val;
      } else {
        int b = r / d, m = r % d;
        int y = n / d, vv = n % d;
        Xact[(long)(b * f_out + y) * ldxa + oxoff + vv * d + m] = (bf16_t)val;
      }
    }
  }
}

// --------------------------- fused SO(3) activation --------------------------
// Y[r,i] = sum_n D2[n,i] * relu( sum_k X[r,k] * DACT[n,k] )   (sqrt2*qw in D2)
// per wave: 16 rows resident (6 A-frags), 11 resident out tiles.
// B fragments double-buffered through LDS: per 32-point chunk the WG stages
// 12 B1 frags (12KB, actOrder layout) + 11 B2 frags (11KB) once, all 8 waves
// read from LDS; global loads for chunk nc+1 are in flight during compute.
__global__ __launch_bounds__(256) void act_fused(const bf16_t* X, const bf16_t* B1f,
                                                 const bf16_t* B2f, bf16_t* Y,
                                                 int R) {
  __shared__ __align__(16) v4u stage[2][1472];       // 2 x 23KB
  __shared__ __align__(16) bf16_t sbuf[8 * 16 * 40]; // relu restage, per-wave
  int lane = threadIdx.x & 31, wave = threadIdx.x >> 5;
  int row0 = blockIdx.x * 128 + wave * 16;
  bf16_t* myl = sbuf + wave * 16 * 40;

  v16bf xa[6];
#pragma unroll
  for (int kt = 0; kt < 6; ++kt) xa[kt] = load_afrag(X, 192, row0, kt * 32, lane);

  v8f out[11];
#pragma unroll
  for (int t = 0; t < 11; ++t) out[t] = vzero8();

  // prologue: stage chunk 0
  {
    const v4u* p1 = (const v4u*)(B1f);
    const v4u* p2 = (const v4u*)(B2f);
#pragma unroll
    for (int i = 0; i < 6; ++i) {
      int idx = threadIdx.x + i * 256;
      if (idx < 1472) stage[0][idx] = (idx < 768) ? p1[idx] : p2[idx - 768];
    }
  }
  __syncthreads();

  int r = (lane >> 4) << 3, c = lane & 15;
  v4u tmp[6];
  for (int nc = 0; nc < 128; ++nc) {
    int cur = nc & 1;
    if (nc < 127) {  // issue next chunk's global loads (overlap with compute)
      const v4u* p1 = (const v4u*)(B1f + (long)(nc + 1) * 6144);
      const v4u* p2 = (const v4u*)(B2f + (long)(nc + 1) * 5632);
#pragma unroll
      for (int i = 0; i < 6; ++i) {
        int idx = threadIdx.x + i * 256;
        if (idx < 1472) tmp[i] = (idx < 768) ? p1[idx] : p2[idx - 768];
      }
    }
    const bf16_t* sb1 = (const bf16_t*)stage[cur];
    const bf16_t* sb2 = sb1 + 6144;
    // GEMM1: s[16x32] over K=192
    v8f s0 = vzero8(), s1 = vzero8();
#pragma unroll
    for (int kt = 0; kt < 6; ++kt) {
      v16bf b0 = lds_bfrag(sb1, (kt * 2 + 0) * 512, lane);
      v16bf b1 = lds_bfrag(sb1, (kt * 2 + 1) * 512, lane);
      s0 = wmma_bf16(xa[kt], b0, s0);
      s1 = wmma_bf16(xa[kt], b1, s1);
    }
    // relu -> bf16 -> LDS (row-major 16 x 32, stride 40 halves)
#pragma unroll
    for (int v = 0; v < 8; ++v) {
      float a0 = s0[v]; a0 = a0 > 0.f ? a0 : 0.f;
      float a1 = s1[v]; a1 = a1 > 0.f ? a1 : 0.f;
      myl[(r + v) * 40 + c] = (bf16_t)a0;
      myl[(r + v) * 40 + 16 + c] = (bf16_t)a1;
    }
    asm volatile("s_wait_dscnt 0" ::: "memory");  // intra-wave RAW on LDS
    v16bf a2;
    {
      int rr = lane & 15, hs = (lane >> 4) * 8;
      FragU f;
      f.u[0] = *(const v4u*)(myl + rr * 40 + hs);
      f.u[1] = *(const v4u*)(myl + rr * 40 + 16 + hs);
      a2 = f.v;
    }
    asm volatile("s_wait_dscnt 0" ::: "memory");  // WAR before next overwrite
    // GEMM2: out += relu_tile x D2 chunk (11 n-tiles, batched 4/4/3)
    v16bf bb[4];
#pragma unroll
    for (int t = 0; t < 4; ++t) bb[t] = lds_bfrag(sb2, t * 512, lane);
#pragma unroll
    for (int t = 0; t < 4; ++t) out[t] = wmma_bf16(a2, bb[t], out[t]);
#pragma unroll
    for (int t = 0; t < 4; ++t) bb[t] = lds_bfrag(sb2, (t + 4) * 512, lane);
#pragma unroll
    for (int t = 0; t < 4; ++t) out[t + 4] = wmma_bf16(a2, bb[t], out[t + 4]);
#pragma unroll
    for (int t = 0; t < 3; ++t) bb[t] = lds_bfrag(sb2, (t + 8) * 512, lane);
#pragma unroll
    for (int t = 0; t < 3; ++t) out[t + 8] = wmma_bf16(a2, bb[t], out[t + 8]);
    // commit next chunk into the other stage buffer
    if (nc < 127) {
#pragma unroll
      for (int i = 0; i < 6; ++i) {
        int idx = threadIdx.x + i * 256;
        if (idx < 1472) stage[1 - cur][idx] = tmp[i];
      }
    }
    __syncthreads();
  }
  int rbase = row0 + ((lane >> 4) << 3);
#pragma unroll
  for (int t = 0; t < 11; ++t) {
    int n = t * 16 + c;
    if (n >= 165) continue;
#pragma unroll
    for (int v = 0; v < 8; ++v)
      Y[(long)(rbase + v) * 192 + n] = (bf16_t)out[t][v];
  }
}

// ------------------------------ head kernels --------------------------------

__global__ void psi2_kernel(const float* Y, const float* wlin, float* psi2) {
  int t = blockIdx.x * blockDim.x + threadIdx.x;
  if (t >= 128 * 128 * 9) return;
  int i = t % 9;
  long xy = t / 9;
  float s = 0.f;
  for (int n = 0; n < 24; ++n) s += Y[n * 9 + i] * wlin[xy * 24 + n];
  psi2[t] = s * (1.f / 24.f);
}

__global__ void s2_head(const float* psi2, const bf16_t* act2, float* out) {
  int t = blockIdx.x * blockDim.x + threadIdx.x;
  if (t >= 64 * 128 * 9) return;
  int mf = t % 9;
  int y = (t / 9) % 128;
  int b = t / (9 * 128);
  int l, m;
  if (mf == 0) { l = 0; m = 0; }
  else if (mf < 4) { l = 1; m = mf - 1; }
  else { l = 2; m = mf - 4; }
  int d = 2 * l + 1;
  const int ox[3] = {0, 1, 10}, op[3] = {0, 1, 4};
  float s = 0.f;
  for (int x = 0; x < 128; ++x)
    for (int u = 0; u < d; ++u)
      s += psi2[(long)(x * 128 + y) * 9 + op[l] + u] *
           (float)act2[(long)(b * 128 + x) * 192 + ox[l] + u * d + m];
  out[t] = s * rsqrtf(128.f * (float)d);
}

// ------------------------------- launch -------------------------------------

extern "C" void kernel_launch(void* const* d_in, const int* in_sizes, int n_in,
                              void* d_out, int out_size, void* d_ws, size_t ws_size,
                              hipStream_t stream) {
  (void)in_sizes; (void)n_in; (void)out_size;
  const float* x_in = (const float*)d_in[0];    // [64,32,455]
  const float* w0 = (const float*)d_in[1];      // [32,64,192]
  const float* w1 = (const float*)d_in[2];      // [64,128,192]
  const float* wlin = (const float*)d_in[3];    // [128,128,24]
  float* out = (float*)d_out;                   // [64,128,9]

  const int DS_[5] = {1, 3, 5, 7, 9};
  const int OXh[5] = {0, 1, 10, 35, 84};
  char* ws = (char*)d_ws;
  size_t off = 0;
  auto alloc = [&](size_t bytes) {
    off = (off + 255) & ~(size_t)255;
    size_t o = off; off += bytes; return o;
  };
  size_t oDK   = alloc(192 * 176 * 2);
  size_t oDKf  = alloc(6 * 12 * 1024);          // NT padded 11 -> 12
  size_t oDACT = alloc((size_t)4096 * 192 * 2);
  size_t oB1f  = alloc((size_t)6 * 256 * 1024);
  size_t oD2   = alloc((size_t)4096 * 176 * 2);
  size_t oB2f  = alloc((size_t)128 * 11 * 1024);
  size_t oY    = alloc(24 * 9 * 4);
  size_t oW0b  = alloc((size_t)2048 * 192 * 2);
  size_t oPsi0 = alloc((size_t)2048 * 176 * 4);
  size_t oP0f[5], oA1[5], oP1f[5], oA2[5];
  for (int l = 0; l < 5; ++l) oP0f[l] = alloc((size_t)4 * DS_[l] * DS_[l] * 1024);
  for (int l = 0; l < 5; ++l) oA1[l] = alloc((size_t)2048 * DS_[l] * DS_[l] * 2);
  size_t oXa1  = alloc((size_t)4096 * 192 * 2);
  size_t oAct1 = alloc((size_t)4096 * 192 * 2);
  size_t oW1b  = alloc((size_t)8192 * 192 * 2);
  size_t oPsi1 = alloc((size_t)8192 * 176 * 4);
  for (int l = 0; l < 5; ++l) oP1f[l] = alloc((size_t)16 * DS_[l] * DS_[l] * 1024);
  for (int l = 0; l < 5; ++l) oA2[l] = alloc((size_t)4096 * DS_[l] * DS_[l] * 2);
  size_t oXa2  = alloc((size_t)8192 * 192 * 2);
  size_t oAct2 = alloc((size_t)8192 * 192 * 2);
  size_t oPsi2 = alloc((size_t)128 * 128 * 9 * 4);
  size_t total = (off + 255) & ~(size_t)255;
  if (total > ws_size) total = ws_size;
  hipMemsetAsync(d_ws, 0, total, stream);

  auto B = [&](size_t o) { return (bf16_t*)(ws + o); };
  auto F = [&](size_t o) { return (float*)(ws + o); };
  auto cdiv = [](long a, long b) { return (int)((a + b - 1) / b); };

  // constants
  gen_DK<<<1, 192, 0, stream>>>(B(oDK));
  gen_DACT<<<cdiv(4096 * 5, 256), 256, 0, stream>>>(B(oDACT), B(oD2));
  gen_Y<<<1, 24, 0, stream>>>(F(oY));
  repackB<<<cdiv(72L * 32, 256), 256, 0, stream>>>(B(oDK), B(oDKf), 192, 165, 176, 6, 12, 0, 0);
  repackB<<<cdiv(1536L * 32, 256), 256, 0, stream>>>(B(oDACT), B(oB1f), 192, 4096, 192, 6, 256, 1, 1);
  repackB<<<cdiv(1408L * 32, 256), 256, 0, stream>>>(B(oD2), B(oB2f), 4096, 176, 176, 128, 11, 0, 0);

  // weights -> bf16
  cvt_f32_bf16<<<cdiv(393216, 256), 256, 0, stream>>>(w0, B(oW0b), 393216);
  cvt_f32_bf16<<<cdiv(1572864, 256), 256, 0, stream>>>(w1, B(oW1b), 1572864);

  // psi0 = (w0 . D_K)/192  -> f32 [2048,176]
  wmma_gemm<<<dim3(16, 3), 256, 0, stream>>>(B(oW0b), 192, B(oDKf), 6, 12, F(oPsi0),
                                             176, 1.f / 192.f, 2048, 176, 0, nullptr,
                                             0, 0, 1, 1);
  // layer-1 so3_linear (per l): out rows (b,m), cols (y,v)
  for (int l = 0; l < 5; ++l) {
    int d = DS_[l];
    repackPsiB<<<cdiv(4L * d * d * 32, 256), 256, 0, stream>>>(F(oPsi0), 176, OXh[l],
                                                               d, 64, d, 4 * d, B(oP0f[l]));
    repackA_lin<<<cdiv(2048L * d * d, 256), 256, 0, stream>>>(x_in, 1, 455, OXh[l],
                                                              d, 32, B(oA1[l]));
    wmma_gemm<<<dim3(cdiv(64 * d, 128), d), 256, 0, stream>>>(
        B(oA1[l]), 32 * d, B(oP0f[l]), d, 4 * d, nullptr, 0,
        rsqrtf(32.f * (float)d), 64 * d, 64 * d, 1, B(oXa1), 192, OXh[l], d, 64);
  }
  // act 1 (R = 64*64)
  act_fused<<<32, 256, 0, stream>>>(B(oXa1), B(oB1f), B(oB2f), B(oAct1), 4096);

  // psi1 = (w1 . D_K)/192
  wmma_gemm<<<dim3(64, 3), 256, 0, stream>>>(B(oW1b), 192, B(oDKf), 6, 12, F(oPsi1),
                                             176, 1.f / 192.f, 8192, 176, 0, nullptr,
                                             0, 0, 1, 1);
  // layer-2 so3_linear
  for (int l = 0; l < 5; ++l) {
    int d = DS_[l];
    repackPsiB<<<cdiv(16L * d * d * 32, 256), 256, 0, stream>>>(F(oPsi1), 176, OXh[l],
                                                                d, 128, 2 * d, 8 * d, B(oP1f[l]));
    repackA_lin<<<cdiv(4096L * d * d, 256), 256, 0, stream>>>(B(oAct1), 0, 192, OXh[l],
                                                              d, 64, B(oA2[l]));
    wmma_gemm<<<dim3(cdiv(64 * d, 128), 2 * d), 256, 0, stream>>>(
        B(oA2[l]), 64 * d, B(oP1f[l]), 2 * d, 8 * d, nullptr, 0,
        rsqrtf(64.f * (float)d), 64 * d, 128 * d, 1, B(oXa2), 192, OXh[l], d, 128);
  }
  // act 2 (R = 64*128)
  act_fused<<<64, 256, 0, stream>>>(B(oXa2), B(oB1f), B(oB2f), B(oAct2), 8192);

  // head
  psi2_kernel<<<cdiv(128 * 128 * 9, 256), 256, 0, stream>>>(F(oY), wlin, F(oPsi2));
  s2_head<<<cdiv(64 * 128 * 9, 256), 256, 0, stream>>>(F(oPsi2), B(oAct2), out);
}
